// GNNLayer_83425444757686
// MI455X (gfx1250) — matrix operand order repaired and checked by reading, not compiled, not run on hardware
//
#include <hip/hip_runtime.h>
#include <hip/hip_bf16.h>

typedef float    v8f  __attribute__((ext_vector_type(8)));
typedef _Float16 v16h __attribute__((ext_vector_type(16)));
typedef _Float16 v8h  __attribute__((ext_vector_type(8)));
typedef _Float16 h4   __attribute__((ext_vector_type(4)));

__device__ __forceinline__ void cvt_store4(_Float16* p, float4 f) {
    h4 h = { (_Float16)f.x, (_Float16)f.y, (_Float16)f.z, (_Float16)f.w };
    *(h4*)p = h;
}

// One 16-row tile: A[16,K] from LDS (f16, padded pitch), B packed in global
// (per (kt,nt): lane L holds column N=nt*16+L%16, 16 K-halfs in table order).
// Produces 4 n-tiles of 16x16 f32 accumulators.
template<int KT>
__device__ __forceinline__ void wmma_rows16(const _Float16* L, int pitch,
                                            const _Float16* __restrict__ Bp,
                                            v8f acc[4], int lid) {
    const int row = lid & 15;
    const int off = (lid < 16) ? 0 : 8;   // A layout: lanes 0-15 K{0..7,16..23}, 16-31 K{8..15,24..31}
#pragma unroll
    for (int kt = 0; kt < KT; ++kt) {
        const int k0 = kt * 32;
        v8h lo = *(const v8h*)(L + row * pitch + k0 + off);
        v8h hi = *(const v8h*)(L + row * pitch + k0 + 16 + off);
        v16h a = __builtin_shufflevector(lo, hi, 0,1,2,3,4,5,6,7,8,9,10,11,12,13,14,15);
#pragma unroll
        for (int nt = 0; nt < 4; ++nt) {
            v16h b = *(const v16h*)(Bp + (((size_t)(kt * 4 + nt)) * 32 + lid) * 16);
            acc[nt] = __builtin_amdgcn_wmma_f32_16x16x32_f16(
                false, a, false, b, (short)0, acc[nt], false, false);
        }
    }
}

__global__ __launch_bounds__(256) void k_zero(float* p, long n) {
    long i = (long)blockIdx.x * blockDim.x + threadIdx.x;
    if (i < n) p[i] = 0.f;
}

// Pack W[64,Kdim] (row-major, f32) into B-fragment order, f16.
__global__ __launch_bounds__(256) void k_pack(const float* __restrict__ W, int Kdim,
                                              _Float16* __restrict__ out, int total) {
    int i = blockIdx.x * 256 + threadIdx.x;
    if (i >= total) return;
    int j    = i & 15;          // element within lane (selects K)
    int lane = (i >> 4) & 31;   // lane: N = lane&15, K half = lane>>4
    int nt   = (i >> 9) & 3;
    int kt   = i >> 11;
    out[i] = (_Float16)W[(size_t)(nt * 16 + (lane & 15)) * Kdim + kt * 32 + ((j & 8) << 1) + ((lane >> 4) << 3) + (j & 7)];
}

__global__ __launch_bounds__(256) void k_count(const int* __restrict__ dst,
                                               float* __restrict__ cnt, int E) {
    int i = blockIdx.x * 256 + threadIdx.x;
    if (i < E) atomicAdd(cnt + dst[i], 1.0f);
}

// msg: m = relu([src_h[src] || efeats] @ Wmsg^T + b); atomic scatter-add into sums[dst]
__global__ __launch_bounds__(256) void k_msg(const float* __restrict__ src_h,
                                             const float* __restrict__ efeats,
                                             const int* __restrict__ src,
                                             const int* __restrict__ dst,
                                             const _Float16* __restrict__ Bp,
                                             const float* __restrict__ bias,
                                             float* __restrict__ sums, int ntiles) {
    constexpr int P = 104;                 // 96 + pad (16B multiple)
    __shared__ _Float16 lds[8 * 16 * P];
    const int wave = threadIdx.x >> 5, lid = threadIdx.x & 31;
    const int tile = blockIdx.x * 8 + wave;
    if (tile >= ntiles) return;
    _Float16* L = lds + wave * 16 * P;
    const int ebase = tile * 16;
    const int row = lid >> 1, sub = lid & 1;     // 2 lanes gather one row
    const int e = ebase + row;
    const int sIdx = src[e];
    const float4* sp = (const float4*)(src_h + (size_t)sIdx * 64);
#pragma unroll
    for (int i = 0; i < 8; ++i) cvt_store4(L + row * P + sub * 32 + i * 4, sp[sub * 8 + i]);
    const float4* ep = (const float4*)(efeats + (size_t)e * 32);
#pragma unroll
    for (int i = 0; i < 4; ++i) cvt_store4(L + row * P + 64 + sub * 16 + i * 4, ep[sub * 4 + i]);
    asm volatile("s_wait_dscnt 0" ::: "memory");

    v8f acc[4] = {};
    wmma_rows16<3>(L, P, Bp, acc, lid);

    const int dval = dst[ebase + (lid & 15)];
    const int n15 = lid & 15, rbase = (lid < 16) ? 0 : 8;
#pragma unroll
    for (int nt = 0; nt < 4; ++nt) {
        const int n = nt * 16 + n15;
        const float bn = bias[n];
#pragma unroll
        for (int r = 0; r < 8; ++r) {
            const int m = rbase + r;
            float v = acc[nt][r] + bn;
            v = v > 0.f ? v : 0.f;
            const int d = __shfl(dval, m, 32);
            atomicAdd(sums + (size_t)d * 64 + n, v);
        }
    }
}

__global__ __launch_bounds__(256) void k_mean(float* __restrict__ sums,
                                              const float* __restrict__ cnt, long n) {
    long i = (long)blockIdx.x * 256 + threadIdx.x;
    if (i >= n) return;
    float c = cnt[i >> 6];
    sums[i] = sums[i] / (c > 1.f ? c : 1.f);
}

// node: h_out = [dst_h || h_neigh] @ Wapply^T + b
__global__ __launch_bounds__(256) void k_node(const float* __restrict__ dst_h,
                                              const float* __restrict__ hneigh,
                                              const _Float16* __restrict__ Bp,
                                              const float* __restrict__ bias,
                                              float* __restrict__ out, int ntiles) {
    constexpr int P = 136;                 // 128 + pad
    __shared__ _Float16 lds[8 * 16 * P];
    const int wave = threadIdx.x >> 5, lid = threadIdx.x & 31;
    const int tile = blockIdx.x * 8 + wave;
    if (tile >= ntiles) return;
    _Float16* L = lds + wave * 16 * P;
    const int nbase = tile * 16;
    const int row = lid >> 1, sub = lid & 1;
    const int node = nbase + row;
    const float4* hp = (const float4*)(dst_h + (size_t)node * 64);
    const float4* gp = (const float4*)(hneigh + (size_t)node * 64);
#pragma unroll
    for (int i = 0; i < 8; ++i) cvt_store4(L + row * P + sub * 32 + i * 4, hp[sub * 8 + i]);
#pragma unroll
    for (int i = 0; i < 8; ++i) cvt_store4(L + row * P + 64 + sub * 32 + i * 4, gp[sub * 8 + i]);
    asm volatile("s_wait_dscnt 0" ::: "memory");

    v8f acc[4] = {};
    wmma_rows16<4>(L, P, Bp, acc, lid);

    const int n15 = lid & 15, rbase = (lid < 16) ? 0 : 8;
#pragma unroll
    for (int nt = 0; nt < 4; ++nt) {
        const int n = nt * 16 + n15;
        const float bn = bias[n];
#pragma unroll
        for (int r = 0; r < 8; ++r)
            out[(size_t)(nbase + rbase + r) * 64 + n] = acc[nt][r] + bn;
    }
}

// edge: eh = [src_h[src] || efeats || h_neigh[dst]] @ Wedge^T + b
__global__ __launch_bounds__(256) void k_edge(const float* __restrict__ src_h,
                                              const float* __restrict__ efeats,
                                              const float* __restrict__ hneigh,
                                              const int* __restrict__ src,
                                              const int* __restrict__ dst,
                                              const _Float16* __restrict__ Bp,
                                              const float* __restrict__ bias,
                                              float* __restrict__ out, int ntiles) {
    constexpr int P = 168;                 // 160 + pad
    __shared__ _Float16 lds[8 * 16 * P];
    const int wave = threadIdx.x >> 5, lid = threadIdx.x & 31;
    const int tile = blockIdx.x * 8 + wave;
    if (tile >= ntiles) return;
    _Float16* L = lds + wave * 16 * P;
    const int ebase = tile * 16;
    const int row = lid >> 1, sub = lid & 1;
    const int e = ebase + row;
    const int sIdx = src[e];
    const int dIdx = dst[e];
    const float4* sp = (const float4*)(src_h + (size_t)sIdx * 64);
#pragma unroll
    for (int i = 0; i < 8; ++i) cvt_store4(L + row * P + sub * 32 + i * 4, sp[sub * 8 + i]);
    const float4* ep = (const float4*)(efeats + (size_t)e * 32);
#pragma unroll
    for (int i = 0; i < 4; ++i) cvt_store4(L + row * P + 64 + sub * 16 + i * 4, ep[sub * 4 + i]);
    const float4* gp = (const float4*)(hneigh + (size_t)dIdx * 64);
#pragma unroll
    for (int i = 0; i < 8; ++i) cvt_store4(L + row * P + 96 + sub * 32 + i * 4, gp[sub * 8 + i]);
    asm volatile("s_wait_dscnt 0" ::: "memory");

    v8f acc[4] = {};
    wmma_rows16<5>(L, P, Bp, acc, lid);

    const int n15 = lid & 15, rbase = (lid < 16) ? 0 : 8;
#pragma unroll
    for (int nt = 0; nt < 4; ++nt) {
        const int n = nt * 16 + n15;
        const float bn = bias[n];
#pragma unroll
        for (int r = 0; r < 8; ++r)
            out[(size_t)(ebase + rbase + r) * 64 + n] = acc[nt][r] + bn;
    }
}

extern "C" void kernel_launch(void* const* d_in, const int* in_sizes, int n_in,
                              void* d_out, int out_size, void* d_ws, size_t ws_size,
                              hipStream_t stream) {
    const float* src_h  = (const float*)d_in[0];
    const float* dst_h  = (const float*)d_in[1];
    const float* efeats = (const float*)d_in[2];
    const int*   src    = (const int*)d_in[3];
    const int*   dst    = (const int*)d_in[4];
    const float* Wm = (const float*)d_in[5];  const float* bm = (const float*)d_in[6];
    const float* Wa = (const float*)d_in[7];  const float* ba = (const float*)d_in[8];
    const float* We = (const float*)d_in[9];  const float* be = (const float*)d_in[10];

    const int N = in_sizes[0] / 64;
    const int E = in_sizes[3];

    // Workspace layout: sums[N*64] f32 | cnt[N] f32 | packed f16 weights (32B aligned)
    float* sums = (float*)d_ws;
    float* cnt  = sums + (size_t)N * 64;
    size_t packOff = ((size_t)N * 65 * sizeof(float) + 31) & ~(size_t)31;
    _Float16* Bmsg   = (_Float16*)((char*)d_ws + packOff);       // 3*2048 halfs
    _Float16* Bapply = Bmsg + 3 * 2048;                          // 4*2048 halfs
    _Float16* Bedge  = Bapply + 4 * 2048;                        // 5*2048 halfs

    float* h_out = (float*)d_out;
    float* eh    = h_out + (size_t)N * 64;

    // 1) zero sums + cnt
    {
        long n = (long)N * 65;
        k_zero<<<(int)((n + 255) / 256), 256, 0, stream>>>(sums, n);
    }
    // 2) pack weights into B-fragment layout (f16)
    k_pack<<<(3 * 2048 + 255) / 256, 256, 0, stream>>>(Wm,  96, Bmsg,   3 * 2048);
    k_pack<<<(4 * 2048 + 255) / 256, 256, 0, stream>>>(Wa, 128, Bapply, 4 * 2048);
    k_pack<<<(5 * 2048 + 255) / 256, 256, 0, stream>>>(We, 160, Bedge,  5 * 2048);
    // 3) edge counts per destination
    k_count<<<(E + 255) / 256, 256, 0, stream>>>(dst, cnt, E);
    // 4) message GEMM + relu + scatter-add  (E multiple of 16 for this harness)
    {
        int tiles = E / 16;
        k_msg<<<(tiles + 7) / 8, 256, 0, stream>>>(src_h, efeats, src, dst, Bmsg, bm, sums, tiles);
    }
    // 5) mean (in place: sums -> h_neigh)
    {
        long n = (long)N * 64;
        k_mean<<<(int)((n + 255) / 256), 256, 0, stream>>>(sums, cnt, n);
    }
    // 6) node GEMM -> h_out
    {
        int tiles = N / 16;
        k_node<<<(tiles + 7) / 8, 256, 0, stream>>>(dst_h, sums, Bapply, ba, h_out, tiles);
    }
    // 7) edge-update GEMM -> eh
    {
        int tiles = E / 16;
        k_edge<<<(tiles + 7) / 8, 256, 0, stream>>>(src_h, efeats, sums, src, dst, Bedge, be, eh, tiles);
    }
}
